// RTNGemmLinear_37082747634167
// MI455X (gfx1250) — compile-verified, matrix operand order checked by reading, not run on hardware
//
#include <hip/hip_runtime.h>

// ---- CDNA5 WMMA vector types ----
typedef _Float16 v16h  __attribute__((ext_vector_type(16)));
typedef _Float16 half8 __attribute__((ext_vector_type(8)));
typedef _Float16 half4 __attribute__((ext_vector_type(4)));
typedef float    v8f   __attribute__((ext_vector_type(8)));

// Problem constants (match reference)
#define M_TOK 4096
#define N_OUT 4096
#define K_IN  4096
#define GROUP 128
#define NGRP  (K_IN / GROUP)

// Tiling: 256 threads = 8 waves; block tile 128x128; K tile 64 (half a quant group)
#define BM 128
#define BN 128
#define BK 64
#define LDSP 72    // LDS row pitch in halfs (144 B): 16B-aligned, bank start rotates 36 dw/row

// 16-bit A/B fragment (16x32) from an LDS tile, row-major, pitch LDSP.
__device__ __forceinline__ v16h load_frag(const _Float16* __restrict__ tile,
                                          int row0, int ksub, int frow, int fko) {
  const _Float16* p = tile + (row0 + frow) * LDSP + ksub + fko;
  half8 lo = *reinterpret_cast<const half8*>(p);        // ds_load_b128
  half8 hi = *reinterpret_cast<const half8*>(p + 16);   // ds_load_b128
  return __builtin_shufflevector(lo, hi,
      0, 1, 2, 3, 4, 5, 6, 7, 8, 9, 10, 11, 12, 13, 14, 15);
}

// Issue next tile's global loads into registers (no consumption -> no wait here).
__device__ __forceinline__ void load_tile(const float* __restrict__ X,
                                          const int*   __restrict__ Q,
                                          const float* __restrict__ S,
                                          int block_m, int block_n, int k0,
                                          int rbase, int c4,
                                          float4 (&xv)[8], int4 (&qv)[8],
                                          float (&sv)[8]) {
  const int g = k0 >> 7;
#pragma unroll
  for (int i = 0; i < 8; ++i) {
    const int r = rbase + 16 * i;
    xv[i] = *reinterpret_cast<const float4*>(
        X + (size_t)(block_m + r) * K_IN + k0 + c4);     // global_load_b128
    qv[i] = *reinterpret_cast<const int4*>(
        Q + (size_t)(block_n + r) * K_IN + k0 + c4);     // global_load_b128
    sv[i] = S[(size_t)(block_n + r) * NGRP + g];         // half-wave uniform
  }
}

// Convert staged registers and commit to LDS (the loadcnt wait lands here).
__device__ __forceinline__ void store_tile(_Float16* __restrict__ Xs,
                                           _Float16* __restrict__ Ws,
                                           int rbase, int c4,
                                           const float4 (&xv)[8],
                                           const int4 (&qv)[8],
                                           const float (&sv)[8]) {
#pragma unroll
  for (int i = 0; i < 8; ++i) {
    const int r = rbase + 16 * i;
    half4 hx;
    hx.x = (_Float16)xv[i].x; hx.y = (_Float16)xv[i].y;
    hx.z = (_Float16)xv[i].z; hx.w = (_Float16)xv[i].w;
    *reinterpret_cast<half4*>(&Xs[r * LDSP + c4]) = hx;  // ds_store_b64
    half4 hw;
    hw.x = (_Float16)((float)(qv[i].x - 8) * sv[i]);
    hw.y = (_Float16)((float)(qv[i].y - 8) * sv[i]);
    hw.z = (_Float16)((float)(qv[i].z - 8) * sv[i]);
    hw.w = (_Float16)((float)(qv[i].w - 8) * sv[i]);
    *reinterpret_cast<half4*>(&Ws[r * LDSP + c4]) = hw;  // ds_store_b64
  }
}

__global__ __launch_bounds__(256)
void rtn_gemm_wmma_f16(const float* __restrict__ X,      // [M_TOK, K_IN] f32
                       const float* __restrict__ S,      // [N_OUT, NGRP] f32
                       const int*   __restrict__ Q,      // [N_OUT, K_IN] int4 codes
                       float*       __restrict__ Out) {  // [M_TOK, N_OUT] f32
  __shared__ _Float16 Xs[BM * LDSP];   // 18432 B
  __shared__ _Float16 Ws[BN * LDSP];   // 18432 B

  const int tid  = threadIdx.x;
  const int wave = tid >> 5;
  const int lane = tid & 31;

  const int block_n = blockIdx.x * BN;
  const int block_m = blockIdx.y * BM;

  // wave -> 32(M) x 64(N) sub-tile
  const int wm = (wave >> 1) * 32;
  const int wn = (wave & 1) * 64;

  // fragment lane decomposition
  const int frow = lane & 15;
  const int fko  = (lane >> 4) << 3;

  // staging: half-wave per row, 256 B contiguous per 16 lanes
  const int rbase = tid >> 4;          // 0..15
  const int c4    = (tid & 15) << 2;   // element offset 0..60

  v8f acc[2][4];
#pragma unroll
  for (int i = 0; i < 2; ++i)
#pragma unroll
    for (int j = 0; j < 4; ++j) acc[i][j] = (v8f){};

  float4 xv[8];
  int4   qv[8];
  float  sv[8];

  // prologue: tile 0 into registers
  load_tile(X, Q, S, block_m, block_n, 0, rbase, c4, xv, qv, sv);

  for (int k0 = 0; k0 < K_IN; k0 += BK) {
    // commit staged tile k0 to LDS (waits on its loads here)
    store_tile(Xs, Ws, rbase, c4, xv, qv, sv);
    __syncthreads();

    // pipeline: issue tile k0+BK global loads; they complete under the WMMA block
    if (k0 + BK < K_IN) {
      load_tile(X, Q, S, block_m, block_n, k0 + BK, rbase, c4, xv, qv, sv);
      // warm L2 for the tile after next (global_prefetch_b8)
      if (k0 + 2 * BK < K_IN) {
        __builtin_prefetch(X + (size_t)(block_m + rbase) * K_IN + k0 + 2 * BK + c4, 0, 0);
        __builtin_prefetch(Q + (size_t)(block_n + rbase) * K_IN + k0 + 2 * BK + c4, 0, 0);
      }
    }

    // compute: 2 K-steps x (2x4 tiles) = 16 WMMAs per barrier pair
#pragma unroll
    for (int ks = 0; ks < BK; ks += 32) {
      v16h a[2], b[4];
      a[0] = load_frag(Xs, wm,      ks, frow, fko);
      a[1] = load_frag(Xs, wm + 16, ks, frow, fko);
#pragma unroll
      for (int t = 0; t < 4; ++t)
        b[t] = load_frag(Ws, wn + 16 * t, ks, frow, fko);
#pragma unroll
      for (int t = 0; t < 4; ++t) {
        acc[0][t] = __builtin_amdgcn_wmma_f32_16x16x32_f16(
            false, a[0], false, b[t], (short)0, acc[0][t], false, false);
        acc[1][t] = __builtin_amdgcn_wmma_f32_16x16x32_f16(
            false, a[1], false, b[t], (short)0, acc[1][t], false, false);
      }
    }

    __syncthreads();
  }

  // writeback: f32 C/D layout (lanes 0-15: M=v, lanes 16-31: M=v+8; N=lane&15)
  const int crow = (lane >> 4) << 3;
  const int ccol = lane & 15;
  const int m0 = block_m + wm + crow;
  const int n0 = block_n + wn + ccol;
#pragma unroll
  for (int i = 0; i < 2; ++i) {
#pragma unroll
    for (int t = 0; t < 4; ++t) {
#pragma unroll
      for (int v = 0; v < 8; ++v) {
        Out[(size_t)(m0 + 16 * i + v) * N_OUT + n0 + 16 * t] = acc[i][t][v];
      }
    }
  }
}

extern "C" void kernel_launch(void* const* d_in, const int* in_sizes, int n_in,
                              void* d_out, int out_size, void* d_ws, size_t ws_size,
                              hipStream_t stream) {
  const float* x      = (const float*)d_in[0];   // [4096, 4096] f32
  const float* scales = (const float*)d_in[1];   // [4096, 32]   f32
  const int*   qw     = (const int*)d_in[2];     // [4096, 4096] int32 codes
  float*       out    = (float*)d_out;           // [4096, 4096] f32

  (void)in_sizes; (void)n_in; (void)out_size; (void)d_ws; (void)ws_size;

  dim3 grid(N_OUT / BN, M_TOK / BM);  // 32 x 32 blocks
  dim3 block(256);
  rtn_gemm_wmma_f16<<<grid, block, 0, stream>>>(x, scales, qw, out);
}
